// TwoLayerGCNWithPooling_2233382994098
// MI455X (gfx1250) — matrix operand order repaired and checked by reading, not compiled, not run on hardware
//
#include <hip/hip_runtime.h>
#include <hip/hip_bf16.h>

typedef __attribute__((ext_vector_type(16))) _Float16 v16h;
typedef __attribute__((ext_vector_type(8)))  float    v8f;

#define FH 128        // feature width (F_IN == F_HID == 128)
#define BG 64         // number of graphs

// ---------------------------------------------------------------------------
// Dense projection  Y[N,128] = X[N,128] @ W[128,128]  via v_wmma_f32_16x16x32_f16
// Block = 256 threads = 8 waves. blockIdx.x = row tile (16 rows).
// Wave w computes the 16x16 tile at columns [16w, 16w+16).
// W is staged into LDS transposed ([n][k]) with +8-half row padding.
// ---------------------------------------------------------------------------
__global__ __launch_bounds__(256)
void gemm_wmma_kernel(const float* __restrict__ X, const float* __restrict__ W,
                      float* __restrict__ Y, int nrows) {
  __shared__ _Float16 Wl[FH * (FH + 8)];   // transposed, padded (34,816 B)
  const int tid = threadIdx.x;
  for (int i = tid; i < FH * FH; i += 256) {
    int k = i >> 7, n = i & 127;           // W row-major [k][n]
    Wl[n * (FH + 8) + k] = (_Float16)W[i];
  }
  __syncthreads();

  const int wave = tid >> 5;               // 0..7 -> column tile
  const int lane = tid & 31;
  const int m0   = blockIdx.x * 16;
  const int M    = m0 + (lane & 15);
  if (M >= nrows) return;                  // nrows is a multiple of 16; uniform

  const float* xrow = X + (size_t)M * FH;
  v8f acc = {};
#pragma unroll
  for (int ks = 0; ks < 4; ++ks) {
    // A fragment: lanes 0-15 hold K 0-15, lanes 16-31 hold K 16-31 (matches B)
    const int kbA = ks * 32 + ((lane < 16) ? 0 : 16);
    v16h a;
#pragma unroll
    for (int j = 0; j < 8; ++j) a[j] = (_Float16)xrow[kbA + j];
#pragma unroll
    for (int j = 0; j < 8; ++j) a[8 + j] = (_Float16)xrow[kbA + 8 + j];
    // B fragment: 32x16, lanes 0-15 hold K 0-15, lanes 16-31 hold K 16-31
    const int kbB = ks * 32 + ((lane < 16) ? 0 : 16);
    const _Float16* wp = &Wl[(wave * 16 + (lane & 15)) * (FH + 8) + kbB];
    v16h b;
#pragma unroll
    for (int j = 0; j < 16; ++j) b[j] = wp[j];
    acc = __builtin_amdgcn_wmma_f32_16x16x32_f16(false, a, false, b,
                                                 (short)0, acc, false, false);
  }
  // C/D layout: VGPR r -> row m0 + r (+8 for lanes 16-31), col = wave*16 + lane%16
  const int rbase = m0 + ((lane < 16) ? 0 : 8);
  const int c     = wave * 16 + (lane & 15);
#pragma unroll
  for (int r = 0; r < 8; ++r)
    Y[(size_t)(rbase + r) * FH + c] = acc[r];
}

// ---------------------------------------------------------------------------
__global__ void fill_kernel(float* __restrict__ p, float v, size_t n) {
  size_t i = (size_t)blockIdx.x * blockDim.x + threadIdx.x;
  if (i < n) p[i] = v;
}

__global__ void deg_accum_kernel(const int* __restrict__ dst,
                                 const float* __restrict__ w,
                                 float* __restrict__ deg, int E) {
  int e = blockIdx.x * blockDim.x + threadIdx.x;
  if (e < E) atomicAdd(&deg[dst[e]], w[e]);
}

__global__ void rsqrt_kernel(float* __restrict__ d, int N) {
  int i = blockIdx.x * blockDim.x + threadIdx.x;
  if (i < N) d[i] = rsqrtf(d[i]);
}

// one wave32 per edge; lane handles 4 contiguous features (b128 load + 4 atomics)
__global__ __launch_bounds__(256)
void edge_agg_kernel(const int* __restrict__ src, const int* __restrict__ dst,
                     const float* __restrict__ w, const float* __restrict__ dinv,
                     const float* __restrict__ xW, float* __restrict__ agg, int E) {
  int wid  = (int)(((size_t)blockIdx.x * blockDim.x + threadIdx.x) >> 5);
  int lane = threadIdx.x & 31;
  if (wid >= E) return;
  int   s    = src[wid], d = dst[wid];
  float norm = dinv[s] * w[wid] * dinv[d];
  if (norm == 0.f) return;                  // masked / dropped edges
  const float4 v = ((const float4*)(xW + (size_t)s * FH))[lane];
  float* ad = agg + (size_t)d * FH + lane * 4;
  atomicAdd(ad + 0, norm * v.x);
  atomicAdd(ad + 1, norm * v.y);
  atomicAdd(ad + 2, norm * v.z);
  atomicAdd(ad + 3, norm * v.w);
}

// h = agg + dinv^2 * xW + bias  (in place in agg)
__global__ void finalize_gcn_kernel(float* __restrict__ agg,
                                    const float* __restrict__ xW,
                                    const float* __restrict__ dinv,
                                    const float* __restrict__ bias, size_t total) {
  size_t i = (size_t)blockIdx.x * blockDim.x + threadIdx.x;
  if (i >= total) return;
  float di = dinv[i >> 7];
  agg[i] = agg[i] + di * di * xW[i] + bias[i & 127];
}

// score = sigmoid(dot(h, p) / ||p||), one wave per node
__global__ __launch_bounds__(256)
void score_kernel(const float* __restrict__ h, const float* __restrict__ p,
                  float* __restrict__ score, int N) {
  int node = (int)(((size_t)blockIdx.x * blockDim.x + threadIdx.x) >> 5);
  int lane = threadIdx.x & 31;
  if (node >= N) return;
  float dp = 0.f, pp = 0.f;
#pragma unroll
  for (int j = lane; j < FH; j += 32) {
    float pv = p[j];
    dp += h[(size_t)node * FH + j] * pv;
    pp += pv * pv;
  }
#pragma unroll
  for (int off = 16; off > 0; off >>= 1) {
    dp += __shfl_down(dp, off, 32);
    pp += __shfl_down(pp, off, 32);
  }
  if (lane == 0) score[node] = 1.f / (1.f + expf(-dp * rsqrtf(pp)));
}

// per-graph descending bitonic sort of (score, idx); lowest index wins ties.
// block = n threads (one graph), dynamic LDS = n * 8 bytes.
__global__ void topk_kernel(const float* __restrict__ score,
                            int* __restrict__ perm, float* __restrict__ vals,
                            int* __restrict__ inv, int n, int k) {
  extern __shared__ unsigned long long key[];
  const int b = blockIdx.x, t = threadIdx.x;
  float s = score[(size_t)b * n + t];
  key[t] = ((unsigned long long)__float_as_uint(s) << 32) |
           (unsigned long long)(unsigned int)(~t);
  __syncthreads();
  for (int size = 2; size <= n; size <<= 1) {
    for (int stride = size >> 1; stride > 0; stride >>= 1) {
      int j = t ^ stride;
      if (j > t) {
        unsigned long long a = key[t], c = key[j];
        bool descend = ((t & size) == 0);   // final merge (size==n) -> descending
        bool swap = descend ? (a < c) : (a > c);
        if (swap) { key[t] = c; key[j] = a; }
      }
      __syncthreads();
    }
  }
  int idx = (int)(~(unsigned int)(key[t] & 0xffffffffull));   // original local idx
  if (t < k) {
    perm[(size_t)b * k + t] = idx;
    vals[(size_t)b * k + t] = __uint_as_float((unsigned int)(key[t] >> 32));
  }
  inv[(size_t)b * n + idx] = (t < k) ? t : -1;
}

// out[b*k+j][f] = relu(h[b*n+perm][f] * val)
__global__ void pool_gather_kernel(const float* __restrict__ h,
                                   const int* __restrict__ perm,
                                   const float* __restrict__ vals,
                                   float* __restrict__ out, int nOld, int k,
                                   size_t total) {
  size_t i = (size_t)blockIdx.x * blockDim.x + threadIdx.x;
  if (i >= total) return;
  int    f  = (int)(i & 127);
  size_t nn = i >> 7;                       // new global node = b*k + j
  int    b  = (int)(nn / k);
  int    old = perm[nn];
  float  x = h[((size_t)b * nOld + old) * FH + f] * vals[nn];
  out[i] = fmaxf(x, 0.f);
}

__global__ void edge_remap_kernel(const int* __restrict__ src,
                                  const int* __restrict__ dst,
                                  const float* __restrict__ w,
                                  const int* __restrict__ inv,
                                  int* __restrict__ src2, int* __restrict__ dst2,
                                  float* __restrict__ w2, int n, int k, int E) {
  int e = (int)((size_t)blockIdx.x * blockDim.x + threadIdx.x);
  if (e >= E) return;
  int s = src[e], d = dst[e];
  int gs = s / n, gd = d / n;
  int sl = inv[(size_t)gs * n + (s - gs * n)];
  int dl = inv[(size_t)gd * n + (d - gd * n)];
  bool keep = (sl >= 0) && (dl >= 0);
  src2[e] = keep ? gs * k + sl : 0;
  dst2[e] = keep ? gd * k + dl : 0;
  w2[e]   = keep ? w[e] : 0.f;
}

// per graph: cat = [max over n nodes || mean over n nodes], 128 threads (one/feature)
__global__ void global_pool_kernel(const float* __restrict__ h,
                                   float* __restrict__ cat, int n) {
  int b = blockIdx.x, f = threadIdx.x;
  float mx = -__builtin_inff(), sm = 0.f;
  for (int j = 0; j < n; ++j) {
    float v = h[((size_t)b * n + j) * FH + f];
    mx = fmaxf(mx, v);
    sm += v;
  }
  cat[(size_t)b * 256 + f]       = mx;
  cat[(size_t)b * 256 + 128 + f] = sm / (float)n;
}

// out[b] = sigmoid(dot(cat[b], Wo) + bo); 256 threads per graph
__global__ void out_head_kernel(const float* __restrict__ cat,
                                const float* __restrict__ Wo,
                                const float* __restrict__ bo,
                                float* __restrict__ out) {
  __shared__ float red[256];
  int b = blockIdx.x, t = threadIdx.x;
  red[t] = cat[(size_t)b * 256 + t] * Wo[t];
  __syncthreads();
  for (int s = 128; s > 0; s >>= 1) {
    if (t < s) red[t] += red[t + s];
    __syncthreads();
  }
  if (t == 0) out[b] = 1.f / (1.f + expf(-(red[0] + bo[0])));
}

// ---------------------------------------------------------------------------
extern "C" void kernel_launch(void* const* d_in, const int* in_sizes, int n_in,
                              void* d_out, int out_size, void* d_ws, size_t ws_size,
                              hipStream_t stream) {
  const float* x    = (const float*)d_in[0];
  const int*   eidx = (const int*)d_in[1];
  const float* ewin = (const float*)d_in[2];
  const float* Wmat[3] = {(const float*)d_in[4], (const float*)d_in[7], (const float*)d_in[10]};
  const float* bvec[3] = {(const float*)d_in[5], (const float*)d_in[8], (const float*)d_in[11]};
  const float* pvec[3] = {(const float*)d_in[6], (const float*)d_in[9], (const float*)d_in[12]};
  const float* Wo = (const float*)d_in[13];
  const float* bo = (const float*)d_in[14];

  const int E  = in_sizes[2];          // 1,048,576 edges (constant through pipeline)
  const int N0 = in_sizes[0] / FH;     // 65,536 total nodes

  // bump allocator on workspace (256B aligned)
  char* ws = (char*)d_ws;
  size_t off = 0;
  auto alloc = [&](size_t bytes) -> void* {
    void* p = ws + off;
    off = (off + bytes + 255) & ~(size_t)255;
    return p;
  };
  float* xW   = (float*)alloc((size_t)N0 * FH * 4);
  float* agg  = (float*)alloc((size_t)N0 * FH * 4);
  float* hcur = (float*)alloc((size_t)N0 * FH * 4);
  float* dinv = (float*)alloc((size_t)N0 * 4);
  float* scr  = (float*)alloc((size_t)N0 * 4);
  int*   perm = (int*)  alloc((size_t)N0 / 2 * 4);
  float* vals = (float*)alloc((size_t)N0 / 2 * 4);
  int*   inv  = (int*)  alloc((size_t)N0 * 4);
  int*   esA  = (int*)  alloc((size_t)E * 4);
  int*   edA  = (int*)  alloc((size_t)E * 4);
  float* ewA  = (float*)alloc((size_t)E * 4);
  int*   esB  = (int*)  alloc((size_t)E * 4);
  int*   edB  = (int*)  alloc((size_t)E * 4);
  float* ewB  = (float*)alloc((size_t)E * 4);

  auto cdiv = [](size_t a, size_t b) { return (unsigned)((a + b - 1) / b); };

  const int*   srcC = eidx;
  const int*   dstC = eidx + E;
  const float* wC   = ewin;
  const float* hin  = x;
  int n = N0 / BG;                     // 1024 nodes per graph initially

  for (int L = 0; L < 3; ++L) {
    const int N = BG * n, k = n / 2;
    const size_t tot = (size_t)N * FH;

    // 1) xW = hin @ W_L   (WMMA)
    gemm_wmma_kernel<<<N / 16, 256, 0, stream>>>(hin, Wmat[L], xW, N);
    // 2) degree: 1 (self loop) + scatter(w) ; dinv = rsqrt(deg)
    fill_kernel<<<cdiv(N, 256), 256, 0, stream>>>(dinv, 1.0f, N);
    deg_accum_kernel<<<cdiv(E, 256), 256, 0, stream>>>(dstC, wC, dinv, E);
    rsqrt_kernel<<<cdiv(N, 256), 256, 0, stream>>>(dinv, N);
    // 3) aggregate normalized neighbor features
    fill_kernel<<<cdiv(tot, 256), 256, 0, stream>>>(agg, 0.0f, tot);
    edge_agg_kernel<<<cdiv((size_t)E * 32, 256), 256, 0, stream>>>(
        srcC, dstC, wC, dinv, xW, agg, E);
    // 4) + self-loop + bias
    finalize_gcn_kernel<<<cdiv(tot, 256), 256, 0, stream>>>(agg, xW, dinv, bvec[L], tot);
    // 5) top-k pooling
    score_kernel<<<cdiv((size_t)N * 32, 256), 256, 0, stream>>>(agg, pvec[L], scr, N);
    topk_kernel<<<BG, n, (size_t)n * 8, stream>>>(scr, perm, vals, inv, n, k);
    const size_t ptot = (size_t)BG * k * FH;
    pool_gather_kernel<<<cdiv(ptot, 256), 256, 0, stream>>>(agg, perm, vals, hcur,
                                                            n, k, ptot);  // fused ReLU
    // 6) re-index edges (not needed after the last layer)
    if (L < 2) {
      int*   s2 = (L == 0) ? esA : esB;
      int*   d2 = (L == 0) ? edA : edB;
      float* w2 = (L == 0) ? ewA : ewB;
      edge_remap_kernel<<<cdiv(E, 256), 256, 0, stream>>>(srcC, dstC, wC, inv,
                                                          s2, d2, w2, n, k, E);
      srcC = s2; dstC = d2; wC = w2;
    }
    hin = hcur;
    n = k;
  }

  // readout: cat = [gmp || gap] -> d_out[64..], out = sigmoid(cat @ Wo + bo) -> d_out[0..63]
  float* out_p = (float*)d_out;
  float* cat_p = out_p + BG;
  global_pool_kernel<<<BG, FH, 0, stream>>>(hcur, cat_p, n);
  out_head_kernel<<<BG, 256, 0, stream>>>(cat_p, Wo, bo, out_p);
}
